// Sent_Posit_Drmm_Modeler_52785148068284
// MI455X (gfx1250) — compile-verified, matrix operand order checked by reading, not run on hardware
//
#include <hip/hip_runtime.h>

typedef __attribute__((ext_vector_type(16))) __bf16 v16bf;
typedef __attribute__((ext_vector_type(8)))  float  v8f;

#define GE   200     // embedding dim
#define GH   200     // hidden dim
#define KP   224     // padded K (7 x 32)
#define SEC  208     // padded gate section (13 x 16)
#define NTILES 39    // 624/16 gate col tiles
#define KTILES 7     // 224/32 K tiles
#define WTILE_ELEMS (NTILES*KTILES*512)   // 139776 bf16 per weight matrix

// ---- GRU kernel LDS layout (dynamic shared, no static LDS in kernel) ----
// 32 batch rows per WG (2 row-tiles), 256 threads (8 waves).
#define FRAG_BF16   (2*KTILES*32*16)      // 7168 bf16 per A-staging buffer
#define XBF_OFF     0u
#define HBF_OFF     14336u
#define XF32_OFF    28672u                // two f32 x-tile buffers (async dest)
#define XF32_SZ     25600u                // 32*200*4
#define HST_OFF     (XF32_OFF + 2*XF32_SZ)        // 79872
#define GRZ_OFF     (HST_OFF + 25600u)            // 105472  (32 x 416 f32)
#define GXN_OFF     (GRZ_OFF + 53248u)            // 158720  (32 x 208 f32)
#define GHN_OFF     (GXN_OFF + 26624u)            // 185344  (32 x 208 f32)
#define GRU_SMEM    (GHN_OFF + 26624u)            // 211968 bytes

__device__ __forceinline__ float sigmoidf_(float x){ return 1.0f/(1.0f + expf(-x)); }
__device__ __forceinline__ float leakyf_(float x){ return x > 0.f ? x : 0.1f*x; }

__device__ __forceinline__ unsigned pack_bf16x2(float a, float b) {
  union { __bf16 h[2]; unsigned u; } P;
  P.h[0] = (__bf16)a; P.h[1] = (__bf16)b;
  return P.u;
}

// ---------------------------------------------------------------------------
// Convert GRU weights f32 -> bf16, padded to (624 x 224) and stored in
// WMMA-B-fragment tile order: tile(nt,kt) = 32 lanes x 16 bf16 contiguous.
// Lane c holds gate column nt*16+(c&15); element e maps to
// k = kt*32 + (p&3)*2 + (c>>4)*8 + (p>>2)*16 + (e&1), p=e>>1  (ISA 7.12.2).
// ---------------------------------------------------------------------------
__global__ void convert_weights_kernel(const float* __restrict__ wih_f,
                                       const float* __restrict__ whh_f,
                                       const float* __restrict__ wih_b,
                                       const float* __restrict__ whh_b,
                                       __bf16* __restrict__ out)
{
  int i = blockIdx.x*256 + threadIdx.x;
  if (i >= 4*WTILE_ELEMS) return;
  int d   = i / WTILE_ELEMS;
  int rem = i % WTILE_ELEMS;
  int tile = rem >> 9;          // /512
  int lane = (rem >> 4) & 31;
  int el   = rem & 15;
  int nt = tile / KTILES, kt = tile % KTILES;
  int n = nt*16 + (lane & 15);
  int half = lane >> 4;
  int p = el >> 1;
  int k = kt*32 + (p&3)*2 + half*8 + (p>>2)*16 + (el&1);
  int sec = n / SEC, within = n % SEC;
  const float* src = (d==0) ? wih_f : (d==1) ? whh_f : (d==2) ? wih_b : whh_b;
  float v = 0.f;
  if (within < 200 && k < 200) v = src[(sec*200 + within)*200 + k];
  out[i] = (__bf16)v;
}

// A fragment in LDS is pre-swizzled: frag[(rt*7+kt)*32 + lane] = 16 contiguous bf16.
__device__ __forceinline__ v16bf load_a_frag(const __bf16* __restrict__ base,
                                             int rt, int kt, int lid)
{
  const v16bf* f = (const v16bf*)base;
  return f[(rt*KTILES + kt)*32 + lid];
}

// C fragment store: VGPR j -> row j (+8 for upper half-wave), col = lane&15.
__device__ __forceinline__ void store_c_frag(v8f c, float* __restrict__ buf,
                                             int stride, int colbase, int lid)
{
  int col = colbase + (lid & 15);
  int r0  = (lid >> 4) * 8;
#pragma unroll
  for (int j = 0; j < 8; ++j) buf[(r0 + j)*stride + col] = c[j];
}

// Async-prefetch one 32x200 f32 x-tile into LDS (double buffered).
__device__ __forceinline__ void issue_x_async(const float* __restrict__ x,
                                              int b0, int B, int T, int t,
                                              int bufsel, int tid)
{
  const unsigned ldsbase = XF32_OFF + (unsigned)bufsel * XF32_SZ;
  for (int i = tid; i < 32*100; i += 256) {       // 100 float-pairs per row
    int r = i / 100, m = i - r*100;
    int b = b0 + r;
    if (b < B) {
      unsigned lds = ldsbase + (unsigned)(r*GE + 2*m)*4u;
      const float* g = x + ((size_t)b*T + t)*GE + 2*m;
      asm volatile("global_load_async_to_lds_b64 %0, %1, off"
                   :: "v"(lds), "v"(g) : "memory");
    }
  }
}

// ---------------------------------------------------------------------------
// Persistent-per-WG GRU: 32 batch rows per workgroup, all T steps.
// Gate col layout: [r:0..207][z:208..415][n:416..623].
// fwd (accumulate=0): conv = leaky(h) + x ; bwd (accumulate=1): conv += leaky(h)
// ---------------------------------------------------------------------------
__global__ __launch_bounds__(256)
void gru_kernel(const float* __restrict__ x, float* __restrict__ conv,
                const __bf16* __restrict__ WihP, const __bf16* __restrict__ WhhP,
                const float* __restrict__ bih, const float* __restrict__ bhh,
                const float* __restrict__ h0,
                int B, int T, int backward, int accumulate)
{
  extern __shared__ char smem[];
  __bf16* xbf    = (__bf16*)(smem + XBF_OFF);   // swizzled A fragments of x_t
  __bf16* hbf    = (__bf16*)(smem + HBF_OFF);   // swizzled A fragments of h
  float*  hstate = (float*)(smem + HST_OFF);    // 32*200
  float*  grz    = (float*)(smem + GRZ_OFF);    // 32*416
  float*  gxn    = (float*)(smem + GXN_OFF);    // 32*208
  float*  ghn    = (float*)(smem + GHN_OFF);    // 32*208

  const int tid  = threadIdx.x;
  const int wave = tid >> 5;
  const int lid  = tid & 31;
  const int b0   = blockIdx.x * 32;

  for (int i = tid; i < FRAG_BF16; i += 256) { xbf[i] = (__bf16)0.f; hbf[i] = (__bf16)0.f; }
  for (int i = tid; i < 32*GH; i += 256) hstate[i] = h0[i % GH];

  issue_x_async(x, b0, B, T, backward ? (T-1) : 0, 0, tid);

  for (int s = 0; s < T; ++s) {
    const int t = backward ? (T-1-s) : s;
    const float* xcur = (const float*)(smem + XF32_OFF + (unsigned)(s & 1)*XF32_SZ);

    asm volatile("s_wait_asynccnt 0x0" ::: "memory");
    __syncthreads();   // x tile ready; prev elementwise complete

    // stage x_t and h into swizzled bf16 A-fragment layout (one dword per pair)
    for (int i = tid; i < 32*100; i += 256) {
      int r = i / 100, m = i - r*100;       // k = 2m, 2m+1
      int b = b0 + r;
      float xa = 0.f, xb_ = 0.f;
      if (b < B) { xa = xcur[r*GE + 2*m]; xb_ = xcur[r*GE + 2*m + 1]; }
      float ha = hstate[r*GH + 2*m], hb_ = hstate[r*GH + 2*m + 1];
      int rt = r >> 4, rr = r & 15;
      int kt = (2*m) >> 5;
      int q  = m & 15;                      // (k>>1)&15
      int half = (q >> 2) & 1;
      int p  = (q & 3) | ((q >> 3) << 2);
      int lane = rr + half*16;
      int dw = ((rt*KTILES + kt)*32 + lane)*8 + p;   // dword index in frag buf
      ((unsigned*)xbf)[dw] = pack_bf16x2(xa, xb_);
      ((unsigned*)hbf)[dw] = pack_bf16x2(ha, hb_);
    }
    __syncthreads();

    // prefetch next step's x tile (overlaps GEMM + elementwise)
    if (s + 1 < T) issue_x_async(x, b0, B, T, backward ? (T-2-s) : (s+1), (s+1)&1, tid);

    // 104 tile jobs: 2 row-tiles x {0..25 rz fused, 26..38 xn, 39..51 hn}
    const v16bf* WihV = (const v16bf*)WihP;
    const v16bf* WhhV = (const v16bf*)WhhP;
    for (int jj = wave; jj < 104; jj += 8) {
      const int rt = jj / 52;
      const int j  = jj % 52;
      v8f c = {0.f,0.f,0.f,0.f,0.f,0.f,0.f,0.f};
      if (j < 26) {
        const int nt = j;
#pragma unroll
        for (int kt = 0; kt < KTILES; ++kt) {
          v16bf a  = load_a_frag(xbf, rt, kt, lid);
          v16bf bm = WihV[(nt*KTILES + kt)*32 + lid];
          c = __builtin_amdgcn_wmma_f32_16x16x32_bf16(false, a, false, bm, (short)0, c, false, false);
        }
#pragma unroll
        for (int kt = 0; kt < KTILES; ++kt) {
          v16bf a  = load_a_frag(hbf, rt, kt, lid);
          v16bf bm = WhhV[(nt*KTILES + kt)*32 + lid];
          c = __builtin_amdgcn_wmma_f32_16x16x32_bf16(false, a, false, bm, (short)0, c, false, false);
        }
        store_c_frag(c, grz + rt*16*(2*SEC), 2*SEC, nt*16, lid);
      } else if (j < 39) {
        const int nt = j;               // 26..38: xn tiles
#pragma unroll
        for (int kt = 0; kt < KTILES; ++kt) {
          v16bf a  = load_a_frag(xbf, rt, kt, lid);
          v16bf bm = WihV[(nt*KTILES + kt)*32 + lid];
          c = __builtin_amdgcn_wmma_f32_16x16x32_bf16(false, a, false, bm, (short)0, c, false, false);
        }
        store_c_frag(c, gxn + rt*16*SEC, SEC, (nt-26)*16, lid);
      } else {
        const int nt = j - 13;          // 26..38: hn tiles
#pragma unroll
        for (int kt = 0; kt < KTILES; ++kt) {
          v16bf a  = load_a_frag(hbf, rt, kt, lid);
          v16bf bm = WhhV[(nt*KTILES + kt)*32 + lid];
          c = __builtin_amdgcn_wmma_f32_16x16x32_bf16(false, a, false, bm, (short)0, c, false, false);
        }
        store_c_frag(c, ghn + rt*16*SEC, SEC, (nt-26)*16, lid);
      }
    }
    __syncthreads();

    // elementwise GRU update + output
    for (int i = tid; i < 32*GE; i += 256) {
      int r = i / GE, e = i - r*GE;
      float hv = hstate[r*GH + e];
      float rg = sigmoidf_(grz[r*(2*SEC) + e]       + bih[e]       + bhh[e]);
      float zg = sigmoidf_(grz[r*(2*SEC) + SEC + e] + bih[200 + e] + bhh[200 + e]);
      float ng = tanhf(gxn[r*SEC + e] + bih[400 + e] + rg*(ghn[r*SEC + e] + bhh[400 + e]));
      float hnew = (1.f - zg)*ng + zg*hv;
      hstate[r*GH + e] = hnew;
      int b = b0 + r;
      if (b < B) {
        float o = leakyf_(hnew);
        float* dst = conv + ((size_t)b*T + t)*GE + e;
        if (accumulate) *dst += o;
        else            *dst  = o + xcur[r*GE + e];
      }
    }
    // next iteration's top barrier orders hstate/gate reuse
  }
}

// ---------------------------------------------------------------------------
// qw = softmax over Lq of (concat(q_ctx,q_idfs) @ qm_W^T + qm_b); also norms.
// ---------------------------------------------------------------------------
__global__ void qw_kernel(const float* __restrict__ q_emb, const float* __restrict__ q_ctx,
                          const float* __restrict__ q_idfs,
                          const float* __restrict__ qmW, const float* __restrict__ qmb,
                          float* __restrict__ qw, float* __restrict__ qn_emb,
                          float* __restrict__ qn_ctx)
{
  __shared__ float s[32];
  int q = threadIdx.x;           // 32 threads
  float n1 = 0.f, n2 = 0.f, dot = 0.f;
  for (int e = 0; e < GE; ++e) {
    float a = q_emb[q*GE + e], b = q_ctx[q*GE + e];
    n1 += a*a; n2 += b*b; dot += b * qmW[e];
  }
  dot += q_idfs[q] * qmW[200] + qmb[0];
  qn_emb[q] = sqrtf(n1); qn_ctx[q] = sqrtf(n2);
  s[q] = dot;
  __syncthreads();
  if (q == 0) {
    float m = s[0];
    for (int i = 1; i < 32; ++i) m = fmaxf(m, s[i]);
    float ex[32]; float sum = 0.f;
    for (int i = 0; i < 32; ++i) { ex[i] = expf(s[i] - m); sum += ex[i]; }
    for (int i = 0; i < 32; ++i) qw[i] = ex[i] / sum;
  }
}

// ---------------------------------------------------------------------------
// Per-sentence scoring: cosine sims, top-5 pooling, MLPs, sigmoid gate.
// One block per sentence b. res[b] in (0,1).
// ---------------------------------------------------------------------------
__global__ __launch_bounds__(256)
void score_kernel(const float* __restrict__ sents, const float* __restrict__ conv,
                  const float* __restrict__ q_emb, const float* __restrict__ q_ctx,
                  const float* __restrict__ qn_emb, const float* __restrict__ qn_ctx,
                  const float* __restrict__ qw, const float* __restrict__ af,
                  const float* __restrict__ q1W, const float* __restrict__ q1b,
                  const float* __restrict__ q2W, const float* __restrict__ q2b,
                  const float* __restrict__ outW, const float* __restrict__ outb,
                  float* __restrict__ res)
{
  extern __shared__ char smem[];
  float* ss     = (float*)smem;        // 64*200
  float* sc     = ss + 64*GE;          // 64*200
  float* norm_s = sc + 64*GE;          // 64
  float* norm_c = norm_s + 64;         // 64
  float* sim_i  = norm_c + 64;         // 64
  float* sim_s  = sim_i + 64;          // 64
  float* sq     = sim_s + 64;          // 200
  float* sqc    = sq + GE;             // 200
  float* lo     = sqc + GE;            // 32
  float* pool   = lo + 32;             // 6

  const int b = blockIdx.x, tid = threadIdx.x;

  for (int i = tid; i < 64*GE; i += 256) {
    ss[i] = sents[(size_t)b*64*GE + i];
    sc[i] = conv[(size_t)b*64*GE + i];
  }
  __syncthreads();

  if (tid < 128) {
    int r = tid & 63; bool isC = tid >= 64;
    const float* row = (isC ? sc : ss) + r*GE;
    float acc = 0.f;
    for (int e = 0; e < GE; ++e) acc += row[e]*row[e];
    float nn = sqrtf(acc);
    if (isC) norm_c[r] = nn; else norm_s[r] = nn;
  }
  __syncthreads();

  for (int q = 0; q < 32; ++q) {
    for (int e = tid; e < GE; e += 256) { sq[e] = q_emb[q*GE + e]; sqc[e] = q_ctx[q*GE + e]; }
    __syncthreads();
    if (tid < 128) {
      int sidx = tid & 63; bool isC = tid >= 64;
      const float* row = (isC ? sc : ss) + sidx*GE;
      const float* qr  = isC ? sqc : sq;
      float acc = 0.f;
      for (int e = 0; e < GE; ++e) acc += row[e]*qr[e];
      float den = isC ? (qn_ctx[q]*norm_c[sidx]) : (qn_emb[q]*norm_s[sidx]);
      float v = acc/den;
      if (isC) sim_s[sidx] = v; else sim_i[sidx] = v;
    }
    __syncthreads();
    if (tid < 3) {
      if (tid == 2) {              // one-hot pool from sim_insens > 0.999
        int cnt = 0;
        for (int i = 0; i < 64; ++i) if (sim_i[i] > 0.999f) cnt++;
        pool[0] = cnt > 0 ? 1.f : 0.f;
        pool[1] = (float)(cnt < 5 ? cnt : 5) / 5.f;
      } else {
        const float* sim = (tid == 0) ? sim_i : sim_s;
        float t5[5] = {-1e30f,-1e30f,-1e30f,-1e30f,-1e30f};
        for (int i = 0; i < 64; ++i) {
          float v = sim[i];
          if (v > t5[4]) {
            t5[4] = v;
            for (int k = 4; k > 0; --k)
              if (t5[k] > t5[k-1]) { float tp = t5[k-1]; t5[k-1] = t5[k]; t5[k] = tp; }
          }
        }
        pool[2 + tid*2] = t5[0];
        pool[3 + tid*2] = (t5[0]+t5[1]+t5[2]+t5[3]+t5[4]) * 0.2f;
      }
    }
    __syncthreads();
    if (tid == 0) {
      float acc2 = q2b[0];
      for (int h = 0; h < 8; ++h) {
        float hh = q1b[h];
        for (int f = 0; f < 6; ++f) hh += pool[f]*q1W[h*6 + f];
        hh = leakyf_(hh);
        acc2 += hh * q2W[h];
      }
      lo[q] = acc2 * qw[q];
    }
    __syncthreads();
  }

  if (tid == 0) {
    float emit = 0.f;
    for (int q = 0; q < 32; ++q) emit += lo[q];
    emit *= (1.f/32.f);
    float a = outb[0] + af[b*3+0]*outW[0] + af[b*3+1]*outW[1] + af[b*3+2]*outW[2] + emit*outW[3];
    res[b] = 1.f/(1.f + expf(-a));
  }
}

__global__ void max_kernel(const float* __restrict__ v, int n, float* __restrict__ out)
{
  __shared__ float sm[256];
  int tid = threadIdx.x;
  float m = -1e30f;
  for (int i = tid; i < n; i += 256) m = fmaxf(m, v[i]);
  sm[tid] = m; __syncthreads();
  for (int s = 128; s > 0; s >>= 1) {
    if (tid < s) sm[tid] = fmaxf(sm[tid], sm[tid + s]);
    __syncthreads();
  }
  if (tid == 0) *out = sm[0];
}

// ---------------------------------------------------------------------------
// Mesh GRUs: m1 (B=16, T=12, E=200 -> MH=10), then m2 (B=1, T=16, D=10).
// ---------------------------------------------------------------------------
__global__ void mesh_kernel(const float* __restrict__ meshes,
                            const float* __restrict__ m1h0, const float* __restrict__ m1Wih,
                            const float* __restrict__ m1Whh, const float* __restrict__ m1bih,
                            const float* __restrict__ m1bhh,
                            const float* __restrict__ m2h0, const float* __restrict__ m2Wih,
                            const float* __restrict__ m2Whh, const float* __restrict__ m2bih,
                            const float* __restrict__ m2bhh,
                            float* __restrict__ out)
{
  __shared__ float h1[16][10];
  __shared__ float xg[16][30], hg[16][30];
  __shared__ float h2[10], xg2[30], hg2[30];
  const int tid = threadIdx.x;
  if (tid < 160) h1[tid/10][tid%10] = m1h0[tid%10];
  __syncthreads();
  for (int t = 0; t < 12; ++t) {
    for (int i = tid; i < 480; i += 256) {
      int n = i/30, g = i%30;
      const float* xr = meshes + ((size_t)n*12 + t)*GE;
      float a = 0.f;
      for (int e = 0; e < GE; ++e) a += m1Wih[g*GE + e]*xr[e];
      float hh = 0.f;
      for (int e = 0; e < 10; ++e) hh += m1Whh[g*10 + e]*h1[n][e];
      xg[n][g] = a + m1bih[g];
      hg[n][g] = hh + m1bhh[g];
    }
    __syncthreads();
    for (int i = tid; i < 160; i += 256) {
      int n = i/10, e = i%10;
      float r = sigmoidf_(xg[n][e] + hg[n][e]);
      float z = sigmoidf_(xg[n][10+e] + hg[n][10+e]);
      float nn = tanhf(xg[n][20+e] + r*hg[n][20+e]);
      h1[n][e] = (1.f - z)*nn + z*h1[n][e];
    }
    __syncthreads();
  }
  if (tid < 10) h2[tid] = m2h0[tid];
  __syncthreads();
  for (int t = 0; t < 16; ++t) {
    if (tid < 30) {
      float a = 0.f, hh = 0.f;
      for (int e = 0; e < 10; ++e) { a += m2Wih[tid*10 + e]*h1[t][e]; hh += m2Whh[tid*10 + e]*h2[e]; }
      xg2[tid] = a + m2bih[tid];
      hg2[tid] = hh + m2bhh[tid];
    }
    __syncthreads();
    if (tid < 10) {
      float r = sigmoidf_(xg2[tid] + hg2[tid]);
      float z = sigmoidf_(xg2[10+tid] + hg2[10+tid]);
      float nn = tanhf(xg2[20+tid] + r*hg2[20+tid]);
      h2[tid] = (1.f - z)*nn + z*h2[tid];
    }
    __syncthreads();
  }
  if (tid < 10) out[tid] = h2[tid];
}

__global__ void final_kernel(const float* __restrict__ good_out, const float* __restrict__ bad_out,
                             const float* __restrict__ doc_gaf, const float* __restrict__ doc_baf,
                             const float* __restrict__ gm, const float* __restrict__ bm,
                             const float* __restrict__ fW, const float* __restrict__ fbias,
                             float* __restrict__ out)
{
  if (threadIdx.x == 0 && blockIdx.x == 0) {
    float fg = fbias[0], fb = fbias[0];
    fg += good_out[0]*fW[0]; fb += bad_out[0]*fW[0];
    for (int i = 0; i < 4;  ++i) { fg += doc_gaf[i]*fW[1+i]; fb += doc_baf[i]*fW[1+i]; }
    for (int i = 0; i < 10; ++i) { fg += gm[i]*fW[5+i];      fb += bm[i]*fW[5+i];      }
    float loss = 1.f + fb - fg; if (loss < 0.f) loss = 0.f;
    out[0] = loss; out[1] = fg; out[2] = fb;
  }
}

// ---------------------------------------------------------------------------
extern "C" void kernel_launch(void* const* d_in, const int* in_sizes, int n_in,
                              void* d_out, int out_size, void* d_ws, size_t ws_size,
                              hipStream_t stream)
{
  (void)in_sizes; (void)n_in; (void)out_size; (void)ws_size;

  const float* ctx_h0    = (const float*)d_in[0];
  const float* ctx_Wih_f = (const float*)d_in[1];
  const float* ctx_Whh_f = (const float*)d_in[2];
  const float* ctx_bih_f = (const float*)d_in[3];
  const float* ctx_bhh_f = (const float*)d_in[4];
  const float* ctx_Wih_b = (const float*)d_in[5];
  const float* ctx_Whh_b = (const float*)d_in[6];
  const float* ctx_bih_b = (const float*)d_in[7];
  const float* ctx_bhh_b = (const float*)d_in[8];
  const float* qm_W  = (const float*)d_in[9];
  const float* qm_b  = (const float*)d_in[10];
  const float* q1_W  = (const float*)d_in[11];
  const float* q1_b  = (const float*)d_in[12];
  const float* q2_W  = (const float*)d_in[13];
  const float* q2_b  = (const float*)d_in[14];
  const float* out_W = (const float*)d_in[15];
  const float* out_b = (const float*)d_in[16];
  const float* final_W = (const float*)d_in[17];
  const float* final_b = (const float*)d_in[18];
  const float* m1_h0  = (const float*)d_in[19];
  const float* m1_Wih = (const float*)d_in[20];
  const float* m1_Whh = (const float*)d_in[21];
  const float* m1_bih = (const float*)d_in[22];
  const float* m1_bhh = (const float*)d_in[23];
  const float* m2_h0  = (const float*)d_in[24];
  const float* m2_Wih = (const float*)d_in[25];
  const float* m2_Whh = (const float*)d_in[26];
  const float* m2_bih = (const float*)d_in[27];
  const float* m2_bhh = (const float*)d_in[28];
  const float* doc1   = (const float*)d_in[29];
  const float* doc2   = (const float*)d_in[30];
  const float* q_emb  = (const float*)d_in[31];
  const float* q_idfs = (const float*)d_in[32];
  const float* sents_gaf = (const float*)d_in[33];
  const float* sents_baf = (const float*)d_in[34];
  const float* doc_gaf = (const float*)d_in[35];
  const float* doc_baf = (const float*)d_in[36];
  const float* good_meshes = (const float*)d_in[37];
  const float* bad_meshes  = (const float*)d_in[38];

  float* out = (float*)d_out;   // [loss, fg, fb, gs[1024], bs[1024]]

  // workspace layout
  __bf16* wbase = (__bf16*)d_ws;
  __bf16* wih_f = wbase;
  __bf16* whh_f = wbase + 1*WTILE_ELEMS;
  __bf16* wih_b = wbase + 2*WTILE_ELEMS;
  __bf16* whh_b = wbase + 3*WTILE_ELEMS;
  float* fbase  = (float*)(wbase + 4*WTILE_ELEMS);
  float* q_ctx  = fbase;            // 32*200
  float* qw     = q_ctx + 6400;     // 32
  float* qn_emb = qw + 32;          // 32
  float* qn_ctx = qn_emb + 32;      // 32
  float* scal   = qn_ctx + 32;      // [0]=good_out [1]=bad_out [2..11]=gm [12..21]=bm
  float* conv   = fbase + 8192;     // 1024*64*200 (52.4 MB, reused per doc)

  const size_t SCORE_SMEM = (size_t)(64*GE*2 + 64*4 + GE*2 + 32 + 8)*4;   // ~105 KB

  convert_weights_kernel<<<(4*WTILE_ELEMS + 255)/256, 256, 0, stream>>>(
      ctx_Wih_f, ctx_Whh_f, ctx_Wih_b, ctx_Whh_b, wih_f);

  // query context GRU (B=1, T=32) -> q_ctx
  gru_kernel<<<1, 256, GRU_SMEM, stream>>>(q_emb, q_ctx, wih_f, whh_f,
      ctx_bih_f, ctx_bhh_f, ctx_h0,        1, 32, 0, 0);
  gru_kernel<<<1, 256, GRU_SMEM, stream>>>(q_emb, q_ctx, wih_b, whh_b,
      ctx_bih_b, ctx_bhh_b, ctx_h0 + 200,  1, 32, 1, 1);
  qw_kernel<<<1, 32, 0, stream>>>(q_emb, q_ctx, q_idfs, qm_W, qm_b, qw, qn_emb, qn_ctx);

  // doc1
  gru_kernel<<<32, 256, GRU_SMEM, stream>>>(doc1, conv, wih_f, whh_f,
      ctx_bih_f, ctx_bhh_f, ctx_h0,       1024, 64, 0, 0);
  gru_kernel<<<32, 256, GRU_SMEM, stream>>>(doc1, conv, wih_b, whh_b,
      ctx_bih_b, ctx_bhh_b, ctx_h0 + 200, 1024, 64, 1, 1);
  score_kernel<<<1024, 256, SCORE_SMEM, stream>>>(doc1, conv, q_emb, q_ctx,
      qn_emb, qn_ctx, qw, sents_gaf, q1_W, q1_b, q2_W, q2_b, out_W, out_b, out + 3);
  max_kernel<<<1, 256, 0, stream>>>(out + 3, 1024, scal + 0);

  // doc2
  gru_kernel<<<32, 256, GRU_SMEM, stream>>>(doc2, conv, wih_f, whh_f,
      ctx_bih_f, ctx_bhh_f, ctx_h0,       1024, 64, 0, 0);
  gru_kernel<<<32, 256, GRU_SMEM, stream>>>(doc2, conv, wih_b, whh_b,
      ctx_bih_b, ctx_bhh_b, ctx_h0 + 200, 1024, 64, 1, 1);
  score_kernel<<<1024, 256, SCORE_SMEM, stream>>>(doc2, conv, q_emb, q_ctx,
      qn_emb, qn_ctx, qw, sents_baf, q1_W, q1_b, q2_W, q2_b, out_W, out_b, out + 3 + 1024);
  max_kernel<<<1, 256, 0, stream>>>(out + 3 + 1024, 1024, scal + 1);

  // mesh GRUs
  mesh_kernel<<<1, 256, 0, stream>>>(good_meshes, m1_h0, m1_Wih, m1_Whh, m1_bih, m1_bhh,
                                     m2_h0, m2_Wih, m2_Whh, m2_bih, m2_bhh, scal + 2);
  mesh_kernel<<<1, 256, 0, stream>>>(bad_meshes,  m1_h0, m1_Wih, m1_Whh, m1_bih, m1_bhh,
                                     m2_h0, m2_Wih, m2_Whh, m2_bih, m2_bhh, scal + 12);

  final_kernel<<<1, 32, 0, stream>>>(scal + 0, scal + 1, doc_gaf, doc_baf,
                                     scal + 2, scal + 12, final_W, final_b, out);
}
